// SelectiveAttention_8933531976161
// MI455X (gfx1250) — matrix operand order repaired and checked by reading, not compiled, not run on hardware
//
#include <hip/hip_runtime.h>
#include <hip/hip_bf16.h>

typedef __attribute__((ext_vector_type(16))) _Float16 v16h;
typedef __attribute__((ext_vector_type(8)))  _Float16 v8h;
typedef __attribute__((ext_vector_type(8)))  float    v8f;

#define TQ 2048
#define TK 2048
#define BB 4
#define HH 16
#define BH 64      // BB*HH
#define DD 64      // head dim
#define CC 1024    // model dim
#define MR 8192    // TQ*BB flattened rows
#define NT 4       // 16x16 N-tiles per wave (wave strip = 16 x 64)

__device__ __forceinline__ v8f wmma_step(v16h a, v16h b, v8f c) {
  return __builtin_amdgcn_wmma_f32_16x16x32_f16(
      /*neg_a=*/false, a, /*neg_b=*/false, b,
      /*c_mod=*/(short)0, c, /*reuse_a=*/false, /*reuse_b=*/false);
}

// A fragment (16x32, f16 row-major source). Lane l<16: rows M=l, K 0-7 & 16-23.
// Lane l+16: rows M=l, K 8-15 & 24-31.  (ISA 7.12.2, 16-bit A 16x32)
__device__ __forceinline__ v16h load_a_f16(const _Float16* A, int lda, int row0,
                                           int k0, int lane) {
  int l = lane & 15, hi = lane >> 4;
  const _Float16* p = A + (size_t)(row0 + l) * lda + k0 + hi * 8;
  v8h lo = *(const v8h*)p;
  v8h hh = *(const v8h*)(p + 16);
  v16h a;
#pragma unroll
  for (int i = 0; i < 8; ++i) { a[i] = lo[i]; a[i + 8] = hh[i]; }
  return a;
}

// Same A fragment but f32 source, converted in-register.
__device__ __forceinline__ v16h load_a_f32(const float* A, int lda, int row0,
                                           int k0, int lane) {
  int l = lane & 15, hi = lane >> 4;
  const float* p = A + (size_t)(row0 + l) * lda + k0 + hi * 8;
  v8f lo = *(const v8f*)p;
  v8f hh = *(const v8f*)(p + 16);
  v16h a;
#pragma unroll
  for (int i = 0; i < 8; ++i) {
    a[i]     = (_Float16)lo[i];
    a[i + 8] = (_Float16)hh[i];
  }
  return a;
}

// B fragment (32x16) from BT stored [N][K] row-major: lane l<16 -> col N=l,
// K 0-7 & 16-23; lane l+16 -> col N=l, K 8-15 & 24-31 (matches A).
__device__ __forceinline__ v16h load_b_T(const _Float16* BT, int ldbt, int col0,
                                         int k0, int lane) {
  int l = lane & 15, hi = lane >> 4;
  const _Float16* p = BT + (size_t)(col0 + l) * ldbt + k0 + hi * 8;
  v8h lo = *(const v8h*)p;
  v8h hh = *(const v8h*)(p + 16);
  v16h b;
#pragma unroll
  for (int i = 0; i < 8; ++i) { b[i] = lo[i]; b[i + 8] = hh[i]; }
  return b;
}

// ---------------- weight transpose + f32->f16 convert ----------------
__global__ void k_wt(const float* __restrict__ W, _Float16* __restrict__ WT) {
  int idx = blockIdx.x * 256 + threadIdx.x;   // idx = n*CC + k
  int n = idx >> 10, k = idx & (CC - 1);
  WT[idx] = (_Float16)W[((size_t)k << 10) + n];
}

// ---------------- projection GEMM: (A[8192,1024] f32) x WT + bias -----------
// Each wave: 16 x 64 strip (4 accumulators, A-fragment reused 4x).
// mode 0: q -> qH[bh][t][d] (scaled); mode 1: k -> kH[bh][t][d];
// mode 2: v -> vTH[bh][d][t]
__global__ void k_proj(const float* __restrict__ A, const _Float16* __restrict__ BT,
                       const float* __restrict__ bias, float scale, int mode,
                       _Float16* __restrict__ out) {
  int lane = threadIdx.x & 31, wave = threadIdx.x >> 5;
  int row0 = (blockIdx.y * 8 + wave) * 16;
  int col0 = blockIdx.x * (16 * NT);
  v8f acc[NT] = {};
  for (int k0 = 0; k0 < CC; k0 += 32) {
    __builtin_prefetch(A + (size_t)(row0 + (lane & 15)) * CC + k0 + 128, 0, 1);
    v16h a = load_a_f32(A, CC, row0, k0, lane);
#pragma unroll
    for (int j = 0; j < NT; ++j) {
      v16h b = load_b_T(BT, CC, col0 + 16 * j, k0, lane);
      acc[j] = wmma_step(a, b, acc[j]);
    }
  }
  int l = lane & 15, hi = lane >> 4;
#pragma unroll
  for (int j = 0; j < NT; ++j) {
    int col = col0 + 16 * j + l;
    float bvs = bias[col];
    int h = col >> 6, d = col & 63;
#pragma unroll
    for (int r = 0; r < 8; ++r) {
      int row = row0 + hi * 8 + r;
      int t = row >> 2, b_ = row & 3;      // row = t*B + b
      int bh = b_ * HH + h;
      float v = (acc[j][r] + bvs) * scale;
      if (mode == 2)
        out[((size_t)bh * DD + d) * TK + t] = (_Float16)v;   // v transposed
      else
        out[((size_t)bh * TQ + t) * DD + d] = (_Float16)v;
    }
  }
}

// ---------------- S = q k^T (per bh), logits -> d_out attn region -----------
__global__ void k_qk(const _Float16* __restrict__ qH, const _Float16* __restrict__ kH,
                     float* __restrict__ attn) {
  int lane = threadIdx.x & 31, wave = threadIdx.x >> 5;
  int bh = blockIdx.z;
  int row0 = (blockIdx.y * 8 + wave) * 16;
  int col0 = blockIdx.x * (16 * NT);
  const _Float16* A  = qH + (size_t)bh * TQ * DD;
  const _Float16* BT = kH + (size_t)bh * TK * DD;
  v8f acc[NT] = {};
#pragma unroll
  for (int k0 = 0; k0 < DD; k0 += 32) {
    v16h a = load_a_f16(A, DD, row0, k0, lane);
#pragma unroll
    for (int j = 0; j < NT; ++j) {
      v16h b = load_b_T(BT, DD, col0 + 16 * j, k0, lane);
      acc[j] = wmma_step(a, b, acc[j]);
    }
  }
  int l = lane & 15, hi = lane >> 4;
  float* C = attn + (size_t)bh * TQ * TK;
#pragma unroll
  for (int j = 0; j < NT; ++j)
#pragma unroll
    for (int r = 0; r < 8; ++r)
      C[(size_t)(row0 + hi * 8 + r) * TK + col0 + 16 * j + l] = acc[j][r];
}

// ---------------- row softmax with key-padding mask, in place ---------------
__global__ void k_softmax(float* __restrict__ attn, const unsigned char* __restrict__ mask) {
  int i = blockIdx.x, bh = blockIdx.y;
  int b_ = bh >> 4;                       // bh = b*H + h
  float* row = attn + ((size_t)bh * TQ + i) * TK;
  const unsigned char* m = mask + (size_t)b_ * TK;
  __shared__ float red[256];
  int tid = threadIdx.x;
  float vals[TK / 256];
  float mx = -INFINITY;
#pragma unroll
  for (int j0 = 0; j0 < TK / 256; ++j0) {
    int j = tid + j0 * 256;
    float v = m[j] ? -INFINITY : row[j];
    vals[j0] = v;
    mx = fmaxf(mx, v);
  }
  red[tid] = mx; __syncthreads();
  for (int s = 128; s > 0; s >>= 1) {
    if (tid < s) red[tid] = fmaxf(red[tid], red[tid + s]);
    __syncthreads();
  }
  mx = red[0]; __syncthreads();
  float sum = 0.f;
#pragma unroll
  for (int j0 = 0; j0 < TK / 256; ++j0) {
    float e = expf(vals[j0] - mx);
    vals[j0] = e;
    sum += e;
  }
  red[tid] = sum; __syncthreads();
  for (int s = 128; s > 0; s >>= 1) {
    if (tid < s) red[tid] += red[tid + s];
    __syncthreads();
  }
  float inv = 1.0f / red[0];
#pragma unroll
  for (int j0 = 0; j0 < TK / 256; ++j0)
    row[tid + j0 * 256] = vals[j0] * inv;
}

// ---------------- x = attn @ v (per bh) -> interH f16 -----------------------
// N = 64 total: one wave covers the full N extent -> attn read exactly once.
__global__ void k_pv(const float* __restrict__ attn, const _Float16* __restrict__ vTH,
                     _Float16* __restrict__ interH) {
  int lane = threadIdx.x & 31, wave = threadIdx.x >> 5;
  int bh = blockIdx.z;
  int row0 = (blockIdx.y * 8 + wave) * 16;
  const float*    A  = attn + (size_t)bh * TQ * TK;
  const _Float16* BT = vTH + (size_t)bh * DD * TK;
  v8f acc[NT] = {};
  for (int k0 = 0; k0 < TK; k0 += 32) {
    __builtin_prefetch(A + (size_t)(row0 + (lane & 15)) * TK + k0 + 128, 0, 1);
    v16h a = load_a_f32(A, TK, row0, k0, lane);
#pragma unroll
    for (int j = 0; j < NT; ++j) {
      v16h b = load_b_T(BT, TK, 16 * j, k0, lane);
      acc[j] = wmma_step(a, b, acc[j]);
    }
  }
  int l = lane & 15, hi = lane >> 4;
  int b_ = bh >> 4, h = bh & 15;
#pragma unroll
  for (int j = 0; j < NT; ++j) {
    int col = h * DD + 16 * j + l;
#pragma unroll
    for (int r = 0; r < 8; ++r) {
      int i = row0 + hi * 8 + r;
      interH[((size_t)i * BB + b_) * CC + col] = (_Float16)acc[j][r];
    }
  }
}

// ---------------- out = inter @ Wo + bo -> d_out x region -------------------
__global__ void k_out(const _Float16* __restrict__ interH, const _Float16* __restrict__ WoT,
                      const float* __restrict__ bo, float* __restrict__ xout) {
  int lane = threadIdx.x & 31, wave = threadIdx.x >> 5;
  int row0 = (blockIdx.y * 8 + wave) * 16;
  int col0 = blockIdx.x * (16 * NT);
  v8f acc[NT] = {};
  for (int k0 = 0; k0 < CC; k0 += 32) {
    __builtin_prefetch(interH + (size_t)(row0 + (lane & 15)) * CC + k0 + 256, 0, 1);
    v16h a = load_a_f16(interH, CC, row0, k0, lane);
#pragma unroll
    for (int j = 0; j < NT; ++j) {
      v16h b = load_b_T(WoT, CC, col0 + 16 * j, k0, lane);
      acc[j] = wmma_step(a, b, acc[j]);
    }
  }
  int l = lane & 15, hi = lane >> 4;
#pragma unroll
  for (int j = 0; j < NT; ++j) {
    int col = col0 + 16 * j + l;
    float bvs = bo[col];
#pragma unroll
    for (int r = 0; r < 8; ++r)
      xout[(size_t)(row0 + hi * 8 + r) * CC + col] = acc[j][r] + bvs;
  }
}

extern "C" void kernel_launch(void* const* d_in, const int* in_sizes, int n_in,
                              void* d_out, int out_size, void* d_ws, size_t ws_size,
                              hipStream_t stream) {
  const float* query = (const float*)d_in[0];
  const float* key   = (const float*)d_in[1];
  const float* value = (const float*)d_in[2];
  const unsigned char* mask = (const unsigned char*)d_in[3];
  const float* Wq = (const float*)d_in[4];
  const float* bq = (const float*)d_in[5];
  const float* Wk = (const float*)d_in[6];
  const float* bk = (const float*)d_in[7];
  const float* Wv = (const float*)d_in[8];
  const float* bv = (const float*)d_in[9];
  const float* Wo = (const float*)d_in[10];
  const float* bo = (const float*)d_in[11];

  float* xout = (float*)d_out;                       // [Tq,B,1024]
  float* attn = xout + (size_t)MR * CC;              // [BH,Tq,Tk]

  char* w = (char*)d_ws;
  _Float16* WqT = (_Float16*)w; w += (size_t)CC * CC * 2;
  _Float16* WkT = (_Float16*)w; w += (size_t)CC * CC * 2;
  _Float16* WvT = (_Float16*)w; w += (size_t)CC * CC * 2;
  _Float16* WoT = (_Float16*)w; w += (size_t)CC * CC * 2;
  _Float16* qH  = (_Float16*)w; w += (size_t)BH * TQ * DD * 2;
  _Float16* kH  = (_Float16*)w; w += (size_t)BH * TK * DD * 2;
  _Float16* vTH = (_Float16*)w; w += (size_t)BH * DD * TK * 2;
  _Float16* interH = (_Float16*)w;                   // [8192,1024] f16

  dim3 blk(256);
  dim3 gw(CC * CC / 256);
  k_wt<<<gw, blk, 0, stream>>>(Wq, WqT);
  k_wt<<<gw, blk, 0, stream>>>(Wk, WkT);
  k_wt<<<gw, blk, 0, stream>>>(Wv, WvT);
  k_wt<<<gw, blk, 0, stream>>>(Wo, WoT);

  dim3 gproj(CC / (16 * NT), MR / 128);
  k_proj<<<gproj, blk, 0, stream>>>(query, WqT, bq, 0.125f, 0, qH);  // 1/sqrt(64)
  k_proj<<<gproj, blk, 0, stream>>>(key,   WkT, bk, 1.0f,   1, kH);
  k_proj<<<gproj, blk, 0, stream>>>(value, WvT, bv, 1.0f,   2, vTH);

  k_qk<<<dim3(TK / (16 * NT), TQ / 128, BH), blk, 0, stream>>>(qH, kH, attn);
  k_softmax<<<dim3(TQ, BH), blk, 0, stream>>>(attn, mask);
  k_pv<<<dim3(1, TQ / 128, BH), blk, 0, stream>>>(attn, vTH, interH);
  k_out<<<gproj, blk, 0, stream>>>(interH, WoT, bo, xout);
}
